// CodeGPTLMHeadModel_20315195310834
// MI455X (gfx1250) — matrix operand order repaired and checked by reading, not compiled, not run on hardware
//
#include <hip/hip_runtime.h>
#include <hip/hip_bf16.h>
#include <math.h>

// ---------------- model constants ----------------
#define L_  12
#define B_  2
#define T_  1024
#define D_  1024
#define H_  16
#define HD_ 64
#define F_  4096
#define V_  50257
#define NR_ (B_*T_)          // 2048 token rows
#define EPS_ 1e-5f

typedef __attribute__((ext_vector_type(16))) __bf16 v16bf;
typedef __attribute__((ext_vector_type(8)))  float  v8f;
typedef __attribute__((ext_vector_type(4)))  unsigned int v4u;
typedef __attribute__((ext_vector_type(8)))  int    v8i;
typedef __attribute__((ext_vector_type(4)))  int    v4i;

// ---------------- bf16 helpers (RNE) ----------------
__device__ __forceinline__ unsigned short bf_bits(float f) {
    unsigned u = __builtin_bit_cast(unsigned, f);
    u += 0x7fffu + ((u >> 16) & 1u);
    return (unsigned short)(u >> 16);
}
__device__ __forceinline__ __bf16 f2bf(float f) {
    return __builtin_bit_cast(__bf16, bf_bits(f));
}

// =====================================================================
// Tensor Data Mover: async copy of a 2-D bf16 tile (rows x cols, row
// stride in elements) from global memory into LDS at byte offset lds_off.
// D# packed per CDNA5 ISA (count=1, type=2,
// data_size=1 (2 bytes), 2-D tile; groups 2/3 zero).
// 6-arg builtin form (clang-23 / therock HIP headers).
// Tracked by TENSORcnt; pair with s_wait_tensorcnt + barrier.
// =====================================================================
__device__ __forceinline__ void tdm_load_bf16_tile(
    unsigned lds_off, const unsigned short* gaddr,
    unsigned stride_elems, unsigned tile_cols, unsigned tile_rows)
{
    unsigned long long ga = (unsigned long long)(const void*)gaddr;
    v4u g0;
    g0[0] = 1u;                                   // count=1 (valid), no gather
    g0[1] = lds_off;                              // lds_addr (bytes)
    g0[2] = (unsigned)(ga & 0xffffffffu);         // global_addr[31:0]
    g0[3] = (unsigned)((ga >> 32) & 0x01ffffffu)  // global_addr[56:32]
          | (2u << 30);                           // type = 2 (image)
    unsigned td0 = stride_elems;                  // tensor_dim0 (OOB bound)
    unsigned td1 = 1u << 20;                      // tensor_dim1 (ample)
    v8i g1;
    g1[0] = (int)(1u << 16);                      // data_size=1 (2B), mask=0
    g1[1] = (int)((td0 & 0xffffu) << 16);         // [31:16]=tensor_dim0 lo
    g1[2] = (int)((td0 >> 16) | ((td1 & 0xffffu) << 16));
    g1[3] = (int)((td1 >> 16) | (tile_cols << 16));   // tile_dim0
    g1[4] = (int)(tile_rows & 0xffffu);               // tile_dim1 (tile_dim2=0)
    g1[5] = (int)stride_elems;                        // tensor_dim0_stride lo
    g1[6] = 0;                                        // stride hi / dim1_stride
    g1[7] = 0;
    v4i gz4 = {0, 0, 0, 0};
    v8i gz8 = {0, 0, 0, 0, 0, 0, 0, 0};
    __builtin_amdgcn_tensor_load_to_lds(g0, g1, gz4, gz4, gz8, 0);
}

// =====================================================================
// embed: x[n,d] = embed[ids[n],d] + pos[n%T,d]
// =====================================================================
__global__ __launch_bounds__(256) void embed_kernel(
    const int* __restrict__ ids, const float* __restrict__ emb,
    const float* __restrict__ pos, float* __restrict__ x)
{
    int n = blockIdx.x;
    int t = n & (T_ - 1);
    int id = ids[n];
    const float* er = emb + (size_t)id * D_;
    const float* pr = pos + (size_t)t * D_;
    float* xr = x + (size_t)n * D_;
    for (int d = threadIdx.x; d < D_; d += 256)
        xr[d] = er[d] + pr[d];
}

// =====================================================================
// RMSNorm over D=1024, output bf16
// =====================================================================
__global__ __launch_bounds__(256) void rms_kernel(
    const float* __restrict__ x, const float* __restrict__ w,
    unsigned short* __restrict__ out)
{
    __shared__ float red[256];
    int n = blockIdx.x, tid = threadIdx.x;
    const float* row = x + (size_t)n * D_;
    float s = 0.f;
    for (int d = tid; d < D_; d += 256) { float v = row[d]; s += v * v; }
    red[tid] = s; __syncthreads();
    for (int off = 128; off > 0; off >>= 1) {
        if (tid < off) red[tid] += red[tid + off];
        __syncthreads();
    }
    float rs = rsqrtf(red[0] * (1.f / D_) + EPS_);
    unsigned short* orow = out + (size_t)n * D_;
    for (int d = tid; d < D_; d += 256)
        orow[d] = bf_bits(w[d] * row[d] * rs);
}

// =====================================================================
// QK head norm: rms over HD=64 per (token, head), output bf16
// =====================================================================
__global__ __launch_bounds__(256) void qknorm_kernel(
    const float* __restrict__ src, const float* __restrict__ w,
    unsigned short* __restrict__ out)
{
    __shared__ float sv[4][64];
    int tid = threadIdx.x;
    int rib = tid >> 6, j = tid & 63;
    long rowId = (long)blockIdx.x * 4 + rib;     // n*H + h
    int n = (int)(rowId >> 4);                   // H = 16
    int h = (int)(rowId & 15);
    size_t base = (size_t)n * D_ + h * HD_;
    float v = src[base + j];
    sv[rib][j] = v * v;
    __syncthreads();
    float s = 0.f;
    for (int i = 0; i < 64; ++i) s += sv[rib][i];
    float rs = rsqrtf(s * (1.f / HD_) + EPS_);
    out[base + j] = bf_bits(w[j] * v * rs);
}

// =====================================================================
// Tiled WMMA GEMM:  C[M,N] = A_bf16[M,K] @ W_f32[K,N]
//   EPI 0: store f32   EPI 1: C += (residual)   EPI 2: bf16(gelu(acc))
// block = 128 threads (4 waves); tile 64x64, K-step 32.
// A tile staged via TDM (pure bf16 copy); W converted f32->bf16 by VALU.
// dynamic LDS layout: As 64x32 bf16 @0 (4096B), Bs 32x72 bf16 @4096.
// =====================================================================
#define GEMM_LDS_BYTES (4096 + 32*72*2)

template <int EPI>
__global__ __launch_bounds__(128) void gemm_kernel(
    const unsigned short* __restrict__ A, const float* __restrict__ W,
    float* __restrict__ Cf, unsigned short* __restrict__ Cb,
    int M, int N, int K)
{
    extern __shared__ char smem[];
    __bf16 (*As)[32] = (__bf16(*)[32])(smem);
    __bf16 (*Bs)[72] = (__bf16(*)[72])(smem + 4096);

    const int tid  = threadIdx.x;
    const int lane = tid & 31;
    const int wave = __builtin_amdgcn_readfirstlane(tid >> 5);
    const int mtile = blockIdx.y * 64;
    const int ntile = blockIdx.x * 64;

    v8f z = {0.f,0.f,0.f,0.f,0.f,0.f,0.f,0.f};
    v8f acc[4]; for (int t = 0; t < 4; ++t) acc[t] = z;

    const int brow = tid >> 2, bcb = (tid & 3) * 16;   // B stage: 32x64

    for (int k0 = 0; k0 < K; k0 += 32) {
        // async TDM: this wave's 16x32 bf16 sub-tile of A  -> As[wave*16..]
        tdm_load_bf16_tile((unsigned)(wave * 16 * 32 * 2),
                           A + (size_t)(mtile + wave * 16) * K + k0,
                           (unsigned)K, 32u, 16u);
        // stage B (fp32 weights -> bf16), ragged-N safe
        {
            const float* src = W + (size_t)(k0 + brow) * N + ntile + bcb;
            for (int i = 0; i < 16; ++i) {
                int n = ntile + bcb + i;
                Bs[brow][bcb + i] = (n < N) ? f2bf(src[i]) : f2bf(0.f);
            }
        }
        if (k0 + 32 < K)
            __builtin_prefetch(W + (size_t)(k0 + 32 + brow) * N + ntile + bcb, 0, 1);
        __builtin_amdgcn_s_wait_tensorcnt(0);
        __syncthreads();

        // A fragment: rows wave*16 .. +15 (contiguous k = 16*(lane/16) + e)
        v16bf afrag;
        {
            int mr = wave * 16 + (lane & 15);
            int kadd = (lane >> 4) << 4;
            for (int e = 0; e < 16; ++e) {
                int k = e + kadd;
                afrag[e] = As[mr][k];
            }
        }
        for (int ns = 0; ns < 4; ++ns) {
            v16bf bfrag;
            int kb = (lane >> 4) * 16;
            int col = ns * 16 + (lane & 15);
            for (int e = 0; e < 16; ++e) bfrag[e] = Bs[kb + e][col];
            acc[ns] = __builtin_amdgcn_wmma_f32_16x16x32_bf16(
                false, afrag, false, bfrag, (short)0, acc[ns], false, false);
        }
        __syncthreads();
    }

    // epilogue (C layout: row = r + 8*(lane>=16), col = lane%16)
    const int lrow = (lane >> 4) * 8;
    const int col0 = lane & 15;
    for (int ns = 0; ns < 4; ++ns) {
        int n = ntile + ns * 16 + col0;
        if (n >= N) continue;
        for (int r = 0; r < 8; ++r) {
            size_t idx = (size_t)(mtile + wave * 16 + lrow + r) * N + n;
            float v = acc[ns][r];
            if (EPI == 0) {
                Cf[idx] = v;
            } else if (EPI == 1) {
                Cf[idx] += v;
            } else {
                float g = 0.5f * v * (1.f + erff(v * 0.70710678f));
                Cb[idx] = bf_bits(g);
            }
        }
    }
}

// =====================================================================
// Flash attention per (batch,head). Q/K tiles staged via TDM (bf16 copy),
// V converted f32->bf16 by VALU. Streaming softmax over 64-key tiles.
// Epilogue fuses per-head gate + value residual; writes bf16.
// grid (T/64, B*H), block 128 (4 waves: each owns 16 query rows).
// dynamic LDS: Qs@0 8192, Ks@8192 8192, Vs@16384 9216, Ss@25600 16640,
//              rowm@42240, rowl@42496, rowa@42752  (total 43008 B)
// =====================================================================
#define ATTN_LDS_BYTES 43008

__global__ __launch_bounds__(128) void attn_kernel(
    const unsigned short* __restrict__ qb, const unsigned short* __restrict__ kb,
    const float* __restrict__ vf, const float* __restrict__ gate,
    unsigned short* __restrict__ attb)
{
    extern __shared__ char smem[];
    __bf16 (*Qs)[64] = (__bf16(*)[64])(smem);
    __bf16 (*Ks)[64] = (__bf16(*)[64])(smem + 8192);
    __bf16 (*Vs)[72] = (__bf16(*)[72])(smem + 16384);
    float  (*Ss)[65] = (float (*)[65])(smem + 25600);
    float* rowm = (float*)(smem + 42240);
    float* rowl = (float*)(smem + 42496);
    float* rowa = (float*)(smem + 42752);

    const int qtile = blockIdx.x;
    const int bh = blockIdx.y;
    const int b = bh / H_, h = bh % H_;
    const int tid = threadIdx.x;
    const int lane = tid & 31;
    const int wave = __builtin_amdgcn_readfirstlane(tid >> 5);
    const float scale = 0.125f; // 1/sqrt(64)

    // async TDM: this wave's 16x64 sub-tile of Q
    tdm_load_bf16_tile((unsigned)(wave * 16 * 64 * 2),
                       qb + (size_t)(b * T_ + qtile * 64 + wave * 16) * D_ + h * HD_,
                       (unsigned)D_, 64u, 16u);
    if (tid < 64) { rowm[tid] = -1e30f; rowl[tid] = 0.f; }

    v8f z = {0.f,0.f,0.f,0.f,0.f,0.f,0.f,0.f};
    v8f oacc[4]; for (int t = 0; t < 4; ++t) oacc[t] = z;
    __builtin_amdgcn_s_wait_tensorcnt(0);
    __syncthreads();

    for (int kt = 0; kt <= qtile; ++kt) {
        // async TDM: this wave's 16x64 sub-tile of K
        tdm_load_bf16_tile((unsigned)(8192 + wave * 16 * 64 * 2),
                           kb + (size_t)(b * T_ + kt * 64 + wave * 16) * D_ + h * HD_,
                           (unsigned)D_, 64u, 16u);
        // stage V tile (f32 -> bf16)
        {
            int row = tid >> 1, cb = (tid & 1) * 32;
            const float* vsrc =
                vf + (size_t)(b * T_ + kt * 64 + row) * D_ + h * HD_ + cb;
            for (int i = 0; i < 32; ++i)
                Vs[row][cb + i] = f2bf(vsrc[i]);
        }
        __builtin_amdgcn_s_wait_tensorcnt(0);
        __syncthreads();

        // S = Q @ K^T over HD=64 (2 WMMA k-steps)
        v8f sacc[4]; for (int t = 0; t < 4; ++t) sacc[t] = z;
        for (int ks = 0; ks < 2; ++ks) {
            v16bf aq;
            int mr = wave * 16 + (lane & 15);
            int kadd = (lane >> 4) << 4;
            for (int e = 0; e < 16; ++e)
                aq[e] = Qs[mr][ks * 32 + e + kadd];
            for (int ns = 0; ns < 4; ++ns) {
                v16bf bk;
                int colk = ns * 16 + (lane & 15);
                int kb2 = (lane >> 4) * 16;
                for (int e = 0; e < 16; ++e)
                    bk[e] = Ks[colk][ks * 32 + kb2 + e];   // K^T: B[k][n]=K[n][k]
                sacc[ns] = __builtin_amdgcn_wmma_f32_16x16x32_bf16(
                    false, aq, false, bk, (short)0, sacc[ns], false, false);
            }
        }
        // scale + causal mask -> LDS scores
        {
            int lrow = (lane >> 4) * 8, col0 = lane & 15;
            for (int ns = 0; ns < 4; ++ns)
                for (int r = 0; r < 8; ++r) {
                    int lr = wave * 16 + lrow + r;
                    int lc = ns * 16 + col0;
                    int qg = qtile * 64 + lr, kg = kt * 64 + lc;
                    float v = sacc[ns][r] * scale;
                    Ss[lr][lc] = (kg <= qg) ? v : -1e30f;
                }
        }
        __syncthreads();

        // online softmax (one thread per query row)
        if (tid < 64) {
            float m_old = rowm[tid], mt = -1e30f;
            for (int j = 0; j < 64; ++j) mt = fmaxf(mt, Ss[tid][j]);
            float m_new = fmaxf(m_old, mt);
            float alpha = __expf(m_old - m_new);
            float s = 0.f;
            for (int j = 0; j < 64; ++j) {
                float p = __expf(Ss[tid][j] - m_new);
                Ss[tid][j] = p;
                s += p;
            }
            rowl[tid] = rowl[tid] * alpha + s;
            rowm[tid] = m_new;
            rowa[tid] = alpha;
        }
        __syncthreads();

        // rescale O, then O += P @ V
        {
            int lrow = (lane >> 4) * 8;
            for (int r = 0; r < 8; ++r) {
                float a = rowa[wave * 16 + lrow + r];
                for (int ns = 0; ns < 4; ++ns) oacc[ns][r] *= a;
            }
        }
        for (int ks = 0; ks < 2; ++ks) {
            v16bf ap;
            int mr = wave * 16 + (lane & 15);
            int kadd = (lane >> 4) << 4;
            for (int e = 0; e < 16; ++e)
                ap[e] = f2bf(Ss[mr][ks * 32 + e + kadd]);
            for (int ns = 0; ns < 4; ++ns) {
                v16bf bv;
                int kb2 = (lane >> 4) * 16;
                int colv = ns * 16 + (lane & 15);
                for (int e = 0; e < 16; ++e)
                    bv[e] = Vs[ks * 32 + kb2 + e][colv];
                oacc[ns] = __builtin_amdgcn_wmma_f32_16x16x32_bf16(
                    false, ap, false, bv, (short)0, oacc[ns], false, false);
            }
        }
        __syncthreads();   // before next tile overwrites Ks/Vs/Ss
    }

    // epilogue: gate * softmax-normalized O + value residual -> bf16
    const float g = gate[h];
    const int lrow = (lane >> 4) * 8, col0 = lane & 15;
    for (int ns = 0; ns < 4; ++ns)
        for (int r = 0; r < 8; ++r) {
            int lr = wave * 16 + lrow + r;
            size_t n = (size_t)(b * T_ + qtile * 64 + lr);
            int c = h * HD_ + ns * 16 + col0;
            float o = g * oacc[ns][r] / rowl[lr] + vf[n * D_ + c];
            attb[n * D_ + c] = bf_bits(o);
        }
}

// =====================================================================
// host orchestration
// =====================================================================
extern "C" void kernel_launch(void* const* d_in, const int* in_sizes, int n_in,
                              void* d_out, int out_size, void* d_ws, size_t ws_size,
                              hipStream_t stream) {
    (void)in_sizes; (void)n_in; (void)out_size; (void)ws_size;

    const int*   ids    = (const int*)  d_in[0];
    const float* embedW = (const float*)d_in[1];
    const float* posW   = (const float*)d_in[2];
    const float* Wq     = (const float*)d_in[3];
    const float* Wk     = (const float*)d_in[4];
    const float* Wv     = (const float*)d_in[5];
    const float* Wo     = (const float*)d_in[6];
    const float* ln1    = (const float*)d_in[7];
    const float* ln2    = (const float*)d_in[8];
    const float* qnw    = (const float*)d_in[9];
    const float* knw    = (const float*)d_in[10];
    const float* gate   = (const float*)d_in[11];
    const float* W1     = (const float*)d_in[12];
    const float* W2     = (const float*)d_in[13];
    const float* lnf    = (const float*)d_in[14];
    const float* Wlm    = (const float*)d_in[15];
    float* out = (float*)d_out;

    // workspace carve-up
    char* ws = (char*)d_ws;
    size_t off = 0;
    float*          xf   = (float*)(ws + off);          off += (size_t)NR_ * D_ * 4;
    unsigned short* hb   = (unsigned short*)(ws + off); off += (size_t)NR_ * D_ * 2;
    float*          qf   = (float*)(ws + off);          off += (size_t)NR_ * D_ * 4;
    float*          kf   = (float*)(ws + off);          off += (size_t)NR_ * D_ * 4;
    float*          vfb  = (float*)(ws + off);          off += (size_t)NR_ * D_ * 4;
    unsigned short* qb   = (unsigned short*)(ws + off); off += (size_t)NR_ * D_ * 2;
    unsigned short* kb   = (unsigned short*)(ws + off); off += (size_t)NR_ * D_ * 2;
    unsigned short* attb = (unsigned short*)(ws + off); off += (size_t)NR_ * D_ * 2;
    unsigned short* hidb = (unsigned short*)(ws + off); off += (size_t)NR_ * F_ * 2;

    const dim3 gD (D_ / 64, NR_ / 64);
    const dim3 gF (F_ / 64, NR_ / 64);
    const dim3 gV ((V_ + 63) / 64, NR_ / 64);
    const dim3 gAtt(T_ / 64, B_ * H_);

    embed_kernel<<<NR_, 256, 0, stream>>>(ids, embedW, posW, xf);

    for (int l = 0; l < L_; ++l) {
        const float* wq = Wq + (size_t)l * D_ * D_;
        const float* wk = Wk + (size_t)l * D_ * D_;
        const float* wv = Wv + (size_t)l * D_ * D_;
        const float* wo = Wo + (size_t)l * D_ * D_;
        const float* w1 = W1 + (size_t)l * D_ * F_;
        const float* w2 = W2 + (size_t)l * F_ * D_;

        // attn block
        rms_kernel<<<NR_, 256, 0, stream>>>(xf, ln1 + (size_t)l * D_, hb);
        gemm_kernel<0><<<gD, 128, GEMM_LDS_BYTES, stream>>>(hb, wq, qf, nullptr, NR_, D_, D_);
        gemm_kernel<0><<<gD, 128, GEMM_LDS_BYTES, stream>>>(hb, wk, kf, nullptr, NR_, D_, D_);
        gemm_kernel<0><<<gD, 128, GEMM_LDS_BYTES, stream>>>(hb, wv, vfb, nullptr, NR_, D_, D_);
        qknorm_kernel<<<(NR_ * H_) / 4, 256, 0, stream>>>(qf, qnw + (size_t)l * HD_, qb);
        qknorm_kernel<<<(NR_ * H_) / 4, 256, 0, stream>>>(kf, knw + (size_t)l * HD_, kb);
        attn_kernel<<<gAtt, 128, ATTN_LDS_BYTES, stream>>>(qb, kb, vfb, gate + (size_t)l * H_, attb);
        gemm_kernel<1><<<gD, 128, GEMM_LDS_BYTES, stream>>>(attb, wo, xf, nullptr, NR_, D_, D_);

        // mlp block
        rms_kernel<<<NR_, 256, 0, stream>>>(xf, ln2 + (size_t)l * D_, hb);
        gemm_kernel<2><<<gF, 128, GEMM_LDS_BYTES, stream>>>(hb, w1, nullptr, hidb, NR_, F_, D_);
        gemm_kernel<1><<<gD, 128, GEMM_LDS_BYTES, stream>>>(hidb, w2, xf, nullptr, NR_, D_, F_);
    }

    // final norm + LM head
    rms_kernel<<<NR_, 256, 0, stream>>>(xf, lnf, hb);
    gemm_kernel<0><<<gV, 128, GEMM_LDS_BYTES, stream>>>(hb, Wlm, out, nullptr, NR_, V_, D_);
}
